// SpatialGRU_11690900980101
// MI455X (gfx1250) — matrix operand order repaired and checked by reading, not compiled, not run on hardware
//
#include <hip/hip_runtime.h>

// ---- problem constants -----------------------------------------------------
#define Bn   256
#define Cn   128
#define L1n  24
#define L2n  24
#define BM   32       // batch rows per block
#define HSn  (L2n+1)  // halo-padded grid stride (25)

typedef __attribute__((ext_vector_type(16))) _Float16 v16h;
typedef __attribute__((ext_vector_type(8)))  float    v8f;

// workspace layout (bytes)
#define HN_FLOATS  ((size_t)(HSn)*(HSn)*Bn*Cn)          // 20,480,000
#define SN_FLOATS  ((size_t)L1n*L2n*Bn*Cn)              // 18,874,368
#define OFF_S      (HN_FLOATS*4)
#define OFF_W3     (OFF_S + SN_FLOATS*4)
#define W3_DW      (64*16*32*8)                         // 262,144 dwords (1 MB)
#define U3_DW      (8*12*32*8)                          // 24,576 dwords
#define OFF_U3     (OFF_W3 + (size_t)W3_DW*4)

#define LDS_A_DW   (2*16*32*8)                          // 8192 dwords = 32 KB (A3 / rh3)
#define LDS_BYTES  (LDS_A_DW*4 + 32*1028*4)             // 32,768 + 131,584 = 164,352

__device__ __forceinline__ unsigned pack2(float a, float b) {
    union { _Float16 h[2]; unsigned u; } x;
    x.h[0] = (_Float16)a; x.h[1] = (_Float16)b; return x.u;
}
__device__ __forceinline__ float sigm(float x) { return 1.0f / (1.0f + __expf(-x)); }

union Frag { v16h v; uint4 q[2]; unsigned u[8]; };

// Wbig(512x1024) = [ W(512x896) | rows 384..511 -> Wij(128x128), rows 0..383 -> 0 ]
__device__ __forceinline__ float wbig(const float* W, const float* Wij, int K, int n) {
    if (n < 896) return W[K * 896 + n];
    int nn = n - 896;
    return (K >= 384) ? Wij[(K - 384) * Cn + nn] : 0.0f;
}

// ---- prep kernels ----------------------------------------------------------
__global__ void zero_f32(float* p, long n) {
    long i = (long)blockIdx.x * 256 + threadIdx.x;
    if (i < n) p[i] = 0.0f;
}

// St[i][j][b][c] = inputs[b][c][i][j]
__global__ void transpose_s(const float* __restrict__ in, float* __restrict__ st) {
    long idx = (long)blockIdx.x * 256 + threadIdx.x;
    if (idx >= (long)SN_FLOATS) return;
    int c = (int)(idx & (Cn - 1));
    long t = idx >> 7;
    int b = (int)(t & (Bn - 1));
    long t2 = t >> 8;
    int j = (int)(t2 % L2n);
    int i = (int)(t2 / L2n);
    st[idx] = in[(((long)b * Cn + c) * L1n + i) * L2n + j];
}

// Fragment-major packed weights:
//   W3[((nt*16+kt)*32+lane)*8+v] = pack(Wbig[kt*32+hi*8+2v+(v>=4?8:0)][nt*16+nl], next K row)
//   U3[((nt*12+kt)*32+lane)*8+v] = pack(U   [kt*32+hi*8+2v+(v>=4?8:0)][nt*16+nl], next K row)
__global__ void pack_weights(const float* __restrict__ W, const float* __restrict__ U,
                             const float* __restrict__ Wij,
                             unsigned* __restrict__ W3, unsigned* __restrict__ U3) {
    int idx = blockIdx.x * 256 + threadIdx.x;
    if (idx < W3_DW) {
        int v    = idx & 7;
        int lane = (idx >> 3) & 31;
        int kt   = (idx >> 8) & 15;
        int nt   = idx >> 12;
        int hi = lane >> 4, nl = lane & 15;
        int K0 = kt * 32 + hi * 8 + 2 * v + (v >= 4 ? 8 : 0);
        int n  = nt * 16 + nl;
        W3[idx] = pack2(wbig(W, Wij, K0, n), wbig(W, Wij, K0 + 1, n));
    } else {
        int r = idx - W3_DW;
        if (r < U3_DW) {
            int v    = r & 7;
            int lane = (r >> 3) & 31;
            int rest = r >> 8;
            int kt = rest % 12, nt = rest / 12;
            int hi = lane >> 4, nl = lane & 15;
            int K0 = kt * 32 + hi * 8 + 2 * v + (v >= 4 ? 8 : 0);
            int n  = nt * 16 + nl;
            U3[r] = pack2(U[K0 * Cn + n], U[(K0 + 1) * Cn + n]);
        }
    }
}

__global__ void copy_out(const float* __restrict__ H, float* __restrict__ out) {
    int t = blockIdx.x * 256 + threadIdx.x;
    if (t < Bn * Cn)
        out[t] = H[((size_t)(L1n * HSn + L2n)) * Bn * Cn + t];
}

// map within-tile K-pair index r (0..15) back to (hi, v) of the fragment slot
__device__ __forceinline__ void r_to_hiv(int r, int& hi, int& v) {
    if (r < 8) { hi = r >> 2; v = r & 3; }
    else       { int rr = r - 8; hi = rr >> 2; v = 4 + (rr & 3); }
}

// ---- one anti-diagonal wavefront step --------------------------------------
__global__ __launch_bounds__(256)
void spatial_gru_wave(const float* __restrict__ St, const float* __restrict__ bias,
                      const unsigned* __restrict__ W3, const unsigned* __restrict__ U3,
                      float* __restrict__ H, int kdiag, int imin) {
    extern __shared__ char smem[];
    unsigned* ldsA = (unsigned*)smem;                 // A3 (fragment-major), later rh3
    float*    pre  = (float*)(smem + LDS_A_DW * 4);   // 32 x 1028 f32

    const int i   = imin + blockIdx.y;
    const int j   = kdiag - i;
    const int b0  = blockIdx.x * BM;
    const int tid = threadIdx.x;
    const int lane = tid & 31, wave = tid >> 5;
    const int nl = lane & 15, hi = lane >> 4;

    const float* Htop  = H + ((size_t)( i      * HSn + (j + 1))) * Bn * Cn;
    const float* Hleft = H + ((size_t)((i + 1) * HSn +  j     )) * Bn * Cn;
    const float* Hdiag = H + ((size_t)( i      * HSn +  j     )) * Bn * Cn;
    float*       Hout  = H + ((size_t)((i + 1) * HSn + (j + 1))) * Bn * Cn;
    const float* Scell = St + ((size_t)(i * L2n + j)) * Bn * Cn;

    // ---- stage q = [h_top|h_left|h_diag|s] directly in fragment-major f16 --
    // A3[((mt*16+kt)*32+lane)*8+v] = pack(Q[mt*16+nl][2*kk], [2*kk+1]),
    //   kk = kt*16 + v + 4*hi + (v>=4?4:0)
    for (int d = tid; d < LDS_A_DW; d += 256) {
        int v  = d & 7;
        int ln = (d >> 3) & 31;
        int kt = (d >> 8) & 15;
        int mt = (d >> 12) & 1;
        int m  = mt * 16 + (ln & 15);
        int kk = kt * 16 + v + 4 * (ln >> 4) + (v >= 4 ? 4 : 0);
        int p  = kk >> 6;              // source part (h_top,h_left,h_diag,s)
        int c0 = (kk & 63) << 1;
        const float* src = (p == 0) ? Htop : (p == 1) ? Hleft : (p == 2) ? Hdiag : Scell;
        int base = (b0 + m) * Cn + c0;
        ldsA[d] = pack2(src[base], src[base + 1]);
    }
    __syncthreads();

    // ---- GEMM1: q(32x512) @ Wbig(512x1024) + bias -> pre -------------------
    for (int t = 0; t < 8; ++t) {
        int nt = wave * 8 + t;
        int n  = nt * 16 + nl;
        const uint4* bptr = (const uint4*)(W3 + ((size_t)(nt * 16) * 32 + lane) * 8);
        const uint4* a0p  = (const uint4*)(ldsA + (size_t)lane * 8);            // mt=0
        const uint4* a1p  = (const uint4*)(ldsA + ((size_t)(16 * 32) + lane) * 8); // mt=1
        if (t < 7)
            __builtin_prefetch((const void*)(bptr + 16 * 64), 0, 3);
        v8f acc0 = {}, acc1 = {};
#pragma unroll 4
        for (int kt = 0; kt < 16; ++kt) {
            Frag b, a0, a1;
            b.q[0]  = bptr[kt * 64];     b.q[1]  = bptr[kt * 64 + 1];
            a0.q[0] = a0p[kt * 64];      a0.q[1] = a0p[kt * 64 + 1];
            a1.q[0] = a1p[kt * 64];      a1.q[1] = a1p[kt * 64 + 1];
            acc0 = __builtin_amdgcn_wmma_f32_16x16x32_f16(
                false, a0.v, false, b.v, (short)0, acc0, false, false);
            acc1 = __builtin_amdgcn_wmma_f32_16x16x32_f16(
                false, a1.v, false, b.v, (short)0, acc1, false, false);
        }
        float bv = bias[n];
#pragma unroll
        for (int r = 0; r < 8; ++r) {
            pre[(r + 8 * hi) * 1028 + n]      = acc0[r] + bv;
            pre[(16 + r + 8 * hi) * 1028 + n] = acc1[r] + bv;
        }
    }
    __syncthreads();

    // ---- fusion: r=sigmoid, rh -> fragment-major f16 LDS, softmax(z) -------
    // rh3[((mt*12+kt)*32+(hi*16+nl))*8+v] for K-pair kk = p*64+cp
    for (int t = tid; t < BM * 64; t += 256) {
        int m = t >> 6, cp = t & 63, c0 = cp * 2;
        int gb = (b0 + m) * Cn + c0;
        float hl0 = Hleft[gb], hl1 = Hleft[gb + 1];
        float ht0 = Htop[gb],  ht1 = Htop[gb + 1];
        float hd0 = Hdiag[gb], hd1 = Hdiag[gb + 1];
        float* pr = pre + m * 1028;
        float rhv[3][2];
        rhv[0][0] = sigm(pr[c0])       * hl0; rhv[0][1] = sigm(pr[c0 + 1])       * hl1;
        rhv[1][0] = sigm(pr[128 + c0]) * ht0; rhv[1][1] = sigm(pr[128 + c0 + 1]) * ht1;
        rhv[2][0] = sigm(pr[256 + c0]) * hd0; rhv[2][1] = sigm(pr[256 + c0 + 1]) * hd1;
        int mt = m >> 4, mnl = m & 15;
#pragma unroll
        for (int p = 0; p < 3; ++p) {
            int kk = p * 64 + cp;
            int kt = kk >> 4, r = kk & 15, fhi, fv;
            r_to_hiv(r, fhi, fv);
            ldsA[((mt * 12 + kt) * 32 + (fhi * 16 + mnl)) * 8 + fv] =
                pack2(rhv[p][0], rhv[p][1]);
        }
#pragma unroll
        for (int cc = c0; cc <= c0 + 1; ++cc) {
            float zi = pr[384 + cc], zl = pr[512 + cc], zt = pr[640 + cc], zd = pr[768 + cc];
            float mx = fmaxf(fmaxf(zi, zl), fmaxf(zt, zd));
            float ei = __expf(zi - mx), el = __expf(zl - mx);
            float et = __expf(zt - mx), ed = __expf(zd - mx);
            float inv = 1.0f / (ei + el + et + ed);
            pr[384 + cc] = ei * inv; pr[512 + cc] = el * inv;
            pr[640 + cc] = et * inv; pr[768 + cc] = ed * inv;
        }
    }
    __syncthreads();

    // ---- GEMM2: rh(32x384) @ U(384x128) -> pre[:, 0:128] -------------------
    {
        int nt = wave;                 // 8 N-tiles, one per wave
        int n  = nt * 16 + nl;
        const uint4* bptr = (const uint4*)(U3 + ((size_t)(nt * 12) * 32 + lane) * 8);
        const uint4* r0p  = (const uint4*)(ldsA + (size_t)lane * 8);             // mt=0
        const uint4* r1p  = (const uint4*)(ldsA + ((size_t)(12 * 32) + lane) * 8); // mt=1
        v8f acc0 = {}, acc1 = {};
#pragma unroll 4
        for (int kt = 0; kt < 12; ++kt) {
            Frag b, a0, a1;
            b.q[0]  = bptr[kt * 64];     b.q[1]  = bptr[kt * 64 + 1];
            a0.q[0] = r0p[kt * 64];      a0.q[1] = r0p[kt * 64 + 1];
            a1.q[0] = r1p[kt * 64];      a1.q[1] = r1p[kt * 64 + 1];
            acc0 = __builtin_amdgcn_wmma_f32_16x16x32_f16(
                false, a0.v, false, b.v, (short)0, acc0, false, false);
            acc1 = __builtin_amdgcn_wmma_f32_16x16x32_f16(
                false, a1.v, false, b.v, (short)0, acc1, false, false);
        }
#pragma unroll
        for (int r = 0; r < 8; ++r) {
            pre[(r + 8 * hi) * 1028 + n]      = acc0[r];
            pre[(16 + r + 8 * hi) * 1028 + n] = acc1[r];
        }
    }
    __syncthreads();

    // ---- combine: h = zl*hl + zt*ht + zd*hd + zi*tanh(sWij + b + rhU) ------
    for (int t = tid; t < BM * Cn; t += 256) {
        int m = t >> 7, c = t & 127;
        float* pr = pre + m * 1028;
        float hcand = tanhf(pr[896 + c] + pr[c]);
        int gb = (b0 + m) * Cn + c;
        Hout[gb] = pr[512 + c] * Hleft[gb] + pr[640 + c] * Htop[gb] +
                   pr[768 + c] * Hdiag[gb] + pr[384 + c] * hcand;
    }
}

// ---- host ------------------------------------------------------------------
extern "C" void kernel_launch(void* const* d_in, const int* in_sizes, int n_in,
                              void* d_out, int out_size, void* d_ws, size_t ws_size,
                              hipStream_t stream) {
    const float* inp  = (const float*)d_in[0];   // (256,128,24,24)
    const float* W    = (const float*)d_in[1];   // (512,896)
    const float* U    = (const float*)d_in[2];   // (384,128)
    const float* bias = (const float*)d_in[3];   // (1024,)
    const float* Wij  = (const float*)d_in[4];   // (128,128)
    float* out = (float*)d_out;

    char* ws = (char*)d_ws;
    float*    H  = (float*)ws;
    float*    St = (float*)(ws + OFF_S);
    unsigned* W3 = (unsigned*)(ws + OFF_W3);
    unsigned* U3 = (unsigned*)(ws + OFF_U3);

    zero_f32<<<(int)((HN_FLOATS + 255) / 256), 256, 0, stream>>>(H, (long)HN_FLOATS);
    transpose_s<<<(int)((SN_FLOATS + 255) / 256), 256, 0, stream>>>(inp, St);
    pack_weights<<<(W3_DW + U3_DW + 255) / 256, 256, 0, stream>>>(W, U, Wij, W3, U3);

    for (int k = 0; k <= L1n + L2n - 2; ++k) {
        int imin = k - (L2n - 1); if (imin < 0) imin = 0;
        int imax = (k < L1n - 1) ? k : (L1n - 1);
        dim3 grid(Bn / BM, imax - imin + 1);
        spatial_gru_wave<<<grid, 256, LDS_BYTES, stream>>>(St, bias, W3, U3, H, k, imin);
    }
    copy_out<<<(Bn * Cn + 255) / 256, 256, 0, stream>>>(H, out);
}
